// LSTM_24953759990083
// MI455X (gfx1250) — compile-verified, matrix operand order checked by reading, not run on hardware
//
#include <hip/hip_runtime.h>

// ---------------------------------------------------------------------------
// Bidirectional 2-layer LSTM + FC for MI455X (gfx1250, wave32, WMMA,
// async global->LDS staging with double buffering)
// ---------------------------------------------------------------------------

typedef __attribute__((ext_vector_type(16))) __bf16 v16bf;
typedef __attribute__((ext_vector_type(8)))  float  v8f;

struct __attribute__((aligned(16))) U4 { unsigned int x, y, z, w; };

union Frag {
    struct { U4 lo, hi; } q;   // two 16B chunks per lane (matches 16-bit A/B layout)
    v16bf b;
};

static constexpr int IN_   = 1880;
static constexpr int KPAD0 = 1888;      // 59 * 32, zero-padded K for layer-0 GEMM
static constexpr int H_    = 256;
static constexpr int T_    = 512;
static constexpr int B_    = 32;
static constexpr int NCLS  = 6;
static constexpr int G4H   = 1024;      // 4*H gate width
static constexpr int MROWS = B_ * T_;   // 16384

#define DI __device__ __forceinline__

DI unsigned short f2bf(float f) {
    union { float f; unsigned int u; } v; v.f = f;
    unsigned int u = v.u;
    return (unsigned short)((u + 0x7FFFu + ((u >> 16) & 1u)) >> 16);  // RNE
}
DI float sigf(float x)      { return 1.0f / (1.0f + __expf(-x)); }
DI float tanh_fast(float x) { return 1.0f - 2.0f / (__expf(2.0f * x) + 1.0f); }

// low 32 bits of a generic pointer to __shared__ = LDS byte offset
DI unsigned lds_off(const void* p) {
    return (unsigned)(unsigned long long)p;
}
// async 16B global -> LDS copy (per lane), tracked by ASYNCcnt
DI void async_copy16(unsigned ldsAddr, const void* gaddr) {
    asm volatile("global_load_async_to_lds_b128 %0, %1, off"
                 :: "v"(ldsAddr), "v"(gaddr) : "memory");
}
DI void wait_async0() {
    asm volatile("s_wait_asynccnt 0" ::: "memory");
}

// ---------------------------------------------------------------------------
// Cast x (B*T, 1880) f32 -> (B*T, 1888) bf16, zero-padded K
// ---------------------------------------------------------------------------
__global__ void cast_x_kernel(const float* __restrict__ x,
                              unsigned short* __restrict__ xbf, long long total) {
    long long i = (long long)blockIdx.x * blockDim.x + threadIdx.x;
    if (i >= total) return;
    int d = (int)(i % KPAD0);
    long long row = i / KPAD0;
    xbf[i] = (d < IN_) ? f2bf(x[row * IN_ + d]) : (unsigned short)0;
}

// ---------------------------------------------------------------------------
// Weight prep: fp32 (1024, K) -> bf16 (1024, Kpad) with gate-interleaved row
// permutation p = w*64 + gate*16 + r  (so wave w owns i/f/g/o of hidden block
// [16w, 16w+16) ), plus combined bias bih+bhh in the same permuted order.
// ---------------------------------------------------------------------------
__global__ __launch_bounds__(256)
void prep_weight_kernel(const float* __restrict__ W,
                        const float* __restrict__ bih, const float* __restrict__ bhh,
                        unsigned short* __restrict__ Wout, float* __restrict__ biasOut,
                        int K, int Kpad) {
    int d = blockIdx.x * 256 + threadIdx.x;
    int p = blockIdx.y;                       // permuted gate column [0,1024)
    int wblk = p >> 6, q = (p >> 4) & 3, r = p & 15;
    int g = q * 256 + wblk * 16 + r;          // original gate row
    if (d < Kpad)
        Wout[(size_t)p * Kpad + d] = (d < K) ? f2bf(W[(size_t)g * K + d]) : (unsigned short)0;
    if (d == 0 && biasOut) biasOut[p] = bih[g] + bhh[g];
}

// ---------------------------------------------------------------------------
// GEMM: C(M,1024) = A(M,Kpad bf16) * W(1024,Kpad bf16)^T + bias
// Block tile 128x64, 256 threads (8 waves), wave tile 32x32 (2x2 WMMA tiles).
// Double-buffered LDS filled by global_load_async_to_lds_b128 (ASYNCcnt),
// 80B row pitch -> conflict-free ds_load_b128 fragment gathers.
// ---------------------------------------------------------------------------
__global__ __launch_bounds__(256)
void gemm_bf16_kernel(const unsigned short* __restrict__ A,
                      const unsigned short* __restrict__ W,
                      const float* __restrict__ bias,
                      float* __restrict__ C, int Kpad) {
    __shared__ unsigned short ldsA[2][128 * 40];   // 128 rows x 32 cols, pitch 40
    __shared__ unsigned short ldsB[2][64 * 40];

    const int tid  = threadIdx.x;
    const int lane = tid & 31, wv = tid >> 5;
    const int half = lane >> 4, l15 = lane & 15;
    const int wave_m = wv & 3, wave_n = wv >> 2;
    const size_t rowBase = (size_t)blockIdx.x * 128;
    const int    colBase = blockIdx.y * 64;

    v8f acc[2][2];
    #pragma unroll
    for (int i = 0; i < 2; ++i)
        #pragma unroll
        for (int j = 0; j < 2; ++j)
            #pragma unroll
            for (int v = 0; v < 8; ++v) acc[i][j][v] = 0.0f;

    // issue async staging of one 32-wide K chunk into buffer `buf`
    auto stage = [&](int kc, int buf) {
        #pragma unroll
        for (int i = 0; i < 2; ++i) {           // A tile: 512 chunks, 2 per thread
            int cid = tid + i * 256;
            int r = cid >> 2, ch = cid & 3;
            const unsigned short* g = A + (rowBase + r) * (size_t)Kpad + kc * 32 + ch * 8;
            async_copy16(lds_off(&ldsA[buf][r * 40 + ch * 8]), g);
        }
        {                                       // B tile: 256 chunks, 1 per thread
            int r = tid >> 2, ch = tid & 3;
            const unsigned short* g = W + (size_t)(colBase + r) * Kpad + kc * 32 + ch * 8;
            async_copy16(lds_off(&ldsB[buf][r * 40 + ch * 8]), g);
        }
    };

    const int nk = Kpad >> 5;
    stage(0, 0);
    wait_async0();
    __syncthreads();

    for (int kc = 0; kc < nk; ++kc) {
        const int buf = kc & 1;
        if (kc + 1 < nk) stage(kc + 1, buf ^ 1);   // overlap next-chunk DMA with WMMA

        Frag a[2], b[2];
        #pragma unroll
        for (int mt = 0; mt < 2; ++mt) {
            int m = wave_m * 32 + mt * 16 + l15;
            int k0 = half * 8;
            a[mt].q.lo = *(const U4*)(&ldsA[buf][m * 40 + k0]);
            a[mt].q.hi = *(const U4*)(&ldsA[buf][m * 40 + k0 + 16]);
        }
        #pragma unroll
        for (int nt = 0; nt < 2; ++nt) {
            int n = wave_n * 32 + nt * 16 + l15;
            int k0 = half * 8;
            b[nt].q.lo = *(const U4*)(&ldsB[buf][n * 40 + k0]);
            b[nt].q.hi = *(const U4*)(&ldsB[buf][n * 40 + k0 + 16]);
        }
        #pragma unroll
        for (int mt = 0; mt < 2; ++mt)
            #pragma unroll
            for (int nt = 0; nt < 2; ++nt)
                acc[mt][nt] = __builtin_amdgcn_wmma_f32_16x16x32_bf16(
                    false, a[mt].b, false, b[nt].b, (short)0, acc[mt][nt], false, false);

        if (kc + 1 < nk) {
            wait_async0();       // my async writes for kc+1 done
            __syncthreads();     // everyone's done; safe to read buf^1 / rewrite buf
        }
    }

    #pragma unroll
    for (int mt = 0; mt < 2; ++mt)
        #pragma unroll
        for (int nt = 0; nt < 2; ++nt) {
            int col = colBase + wave_n * 32 + nt * 16 + l15;
            float bv = bias[col];
            #pragma unroll
            for (int v = 0; v < 8; ++v) {
                size_t row = rowBase + wave_m * 32 + mt * 16 + half * 8 + v;
                C[row * G4H + col] = acc[mt][nt][v] + bv;
            }
        }
}

// ---------------------------------------------------------------------------
// Recurrent scan: one workgroup per direction (grid.x = 2), 512 thr = 16 waves.
// Per step: gates(32x1024) = h(32x256 bf16, LDS) @ Whh^T(256x1024 bf16, L2)
// via WMMA; wave w owns permuted columns [64w, 64w+64) = i/f/g/o of hidden
// block [16w, 16w+16)  ->  LSTM elementwise update fully in registers.
// ---------------------------------------------------------------------------
__global__ __launch_bounds__(512)
void lstm_scan_kernel(const float* __restrict__ preF, const float* __restrict__ preB,
                      const unsigned short* __restrict__ whhF,
                      const unsigned short* __restrict__ whhB,
                      unsigned short* __restrict__ outBf, float* __restrict__ outF) {
    const int dir = blockIdx.x;
    const float* pre          = dir ? preB : preF;
    const unsigned short* whh = dir ? whhB : whhF;

    __shared__ unsigned short hsh[32 * 264];   // h state, bf16, pitch 264 (528B)

    const int tid = threadIdx.x, lane = tid & 31, wv = tid >> 5;
    const int half = lane >> 4, l15 = lane & 15;
    const int hidx = wv * 16 + l15;            // this lane's hidden unit
    const int outCol = dir * H_ + hidx;

    for (int i = tid; i < 32 * 264; i += 512) hsh[i] = 0;
    v8f c[2];
    #pragma unroll
    for (int v = 0; v < 8; ++v) { c[0][v] = 0.0f; c[1][v] = 0.0f; }
    __syncthreads();

    for (int s = 0; s < T_; ++s) {
        const int t = dir ? (T_ - 1 - s) : s;

        v8f acc[2][4];
        #pragma unroll
        for (int mt = 0; mt < 2; ++mt)
            #pragma unroll
            for (int q = 0; q < 4; ++q)
                #pragma unroll
                for (int v = 0; v < 8; ++v) acc[mt][q][v] = 0.0f;

        #pragma unroll
        for (int kc = 0; kc < 8; ++kc) {       // K = 256, chunks of 32
            Frag a[2];
            #pragma unroll
            for (int mt = 0; mt < 2; ++mt) {
                int m = mt * 16 + l15;
                int k0 = kc * 32 + half * 8;
                a[mt].q.lo = *(const U4*)(&hsh[m * 264 + k0]);
                a[mt].q.hi = *(const U4*)(&hsh[m * 264 + k0 + 16]);
            }
            Frag b[4];
            #pragma unroll
            for (int q = 0; q < 4; ++q) {
                int n = wv * 64 + q * 16 + l15;
                const unsigned short* wp = whh + (size_t)n * H_ + kc * 32 + half * 8;
                b[q].q.lo = *(const U4*)(wp);
                b[q].q.hi = *(const U4*)(wp + 16);
            }
            #pragma unroll
            for (int mt = 0; mt < 2; ++mt)
                #pragma unroll
                for (int q = 0; q < 4; ++q)
                    acc[mt][q] = __builtin_amdgcn_wmma_f32_16x16x32_bf16(
                        false, a[mt].b, false, b[q].b, (short)0, acc[mt][q], false, false);
        }
        __syncthreads();   // all hsh reads for this step done

        #pragma unroll
        for (int mt = 0; mt < 2; ++mt) {
            #pragma unroll
            for (int v = 0; v < 8; ++v) {
                int bb = mt * 16 + half * 8 + v;                       // batch row
                size_t prow = ((size_t)bb * T_ + t) * G4H + wv * 64 + l15;
                float gi = acc[mt][0][v] + pre[prow +  0];
                float gf = acc[mt][1][v] + pre[prow + 16];
                float gg = acc[mt][2][v] + pre[prow + 32];
                float go = acc[mt][3][v] + pre[prow + 48];
                float iv = sigf(gi), fv = sigf(gf);
                float gv = tanh_fast(gg), ov = sigf(go);
                float cv = fv * c[mt][v] + iv * gv;
                c[mt][v] = cv;
                float hv = ov * tanh_fast(cv);
                hsh[bb * 264 + hidx] = f2bf(hv);
                size_t oidx = ((size_t)bb * T_ + t) * (size_t)(2 * H_) + outCol;
                if (outBf) outBf[oidx] = f2bf(hv);
                else       outF[oidx]  = hv;
            }
        }
        __syncthreads();   // h(t) visible before next step's reads
    }
}

// ---------------------------------------------------------------------------
// FC: out(bt, c) = h2(bt, :) . Wfc(c, :) + bfc(c)   (tiny, N=6)
// ---------------------------------------------------------------------------
__global__ __launch_bounds__(256)
void fc_kernel(const float* __restrict__ h2, const float* __restrict__ Wfc,
               const float* __restrict__ bfc, float* __restrict__ out) {
    int idx = blockIdx.x * 256 + threadIdx.x;
    if (idx >= MROWS * NCLS) return;
    int bt = idx / NCLS, cc = idx % NCLS;
    const float* hr = h2  + (size_t)bt * 512;
    const float* wr = Wfc + (size_t)cc * 512;
    float s = bfc[cc];
    for (int h = 0; h < 512; ++h) s = fmaf(hr[h], wr[h], s);
    out[idx] = s;
}

// ---------------------------------------------------------------------------
extern "C" void kernel_launch(void* const* d_in, const int* in_sizes, int n_in,
                              void* d_out, int out_size, void* d_ws, size_t ws_size,
                              hipStream_t stream) {
    (void)in_sizes; (void)n_in; (void)out_size; (void)ws_size;

    const float* x     = (const float*)d_in[0];
    const float* Wih0f = (const float*)d_in[1];
    const float* Whh0f = (const float*)d_in[2];
    const float* bih0f = (const float*)d_in[3];
    const float* bhh0f = (const float*)d_in[4];
    const float* Wih0b = (const float*)d_in[5];
    const float* Whh0b = (const float*)d_in[6];
    const float* bih0b = (const float*)d_in[7];
    const float* bhh0b = (const float*)d_in[8];
    const float* Wih1f = (const float*)d_in[9];
    const float* Whh1f = (const float*)d_in[10];
    const float* bih1f = (const float*)d_in[11];
    const float* bhh1f = (const float*)d_in[12];
    const float* Wih1b = (const float*)d_in[13];
    const float* Whh1b = (const float*)d_in[14];
    const float* bih1b = (const float*)d_in[15];
    const float* bhh1b = (const float*)d_in[16];
    const float* Wfc   = (const float*)d_in[17];
    const float* bfc   = (const float*)d_in[18];

    char* ws = (char*)d_ws;
    size_t off = 0;
    auto alloc = [&](size_t bytes) -> char* {
        char* p = ws + off;
        off += (bytes + 255) & ~(size_t)255;
        return p;
    };

    unsigned short* xbf    = (unsigned short*)alloc((size_t)MROWS * KPAD0 * 2);
    unsigned short* w0f    = (unsigned short*)alloc((size_t)G4H * KPAD0 * 2);
    unsigned short* w0b    = (unsigned short*)alloc((size_t)G4H * KPAD0 * 2);
    unsigned short* w1f    = (unsigned short*)alloc((size_t)G4H * 512 * 2);
    unsigned short* w1b    = (unsigned short*)alloc((size_t)G4H * 512 * 2);
    unsigned short* whh0f  = (unsigned short*)alloc((size_t)G4H * H_ * 2);
    unsigned short* whh0b  = (unsigned short*)alloc((size_t)G4H * H_ * 2);
    unsigned short* whh1f  = (unsigned short*)alloc((size_t)G4H * H_ * 2);
    unsigned short* whh1b  = (unsigned short*)alloc((size_t)G4H * H_ * 2);
    float*          bias0f = (float*)alloc(G4H * 4);
    float*          bias0b = (float*)alloc(G4H * 4);
    float*          bias1f = (float*)alloc(G4H * 4);
    float*          bias1b = (float*)alloc(G4H * 4);
    float*          preF   = (float*)alloc((size_t)MROWS * G4H * 4);
    float*          preB   = (float*)alloc((size_t)MROWS * G4H * 4);
    unsigned short* h1bf   = (unsigned short*)alloc((size_t)MROWS * 512 * 2);
    float*          h2     = (float*)alloc((size_t)MROWS * 512 * 4);

    // pass-through output: x
    hipMemcpyAsync(d_out, x, (size_t)MROWS * IN_ * sizeof(float),
                   hipMemcpyDeviceToDevice, stream);

    long long totX = (long long)MROWS * KPAD0;
    cast_x_kernel<<<(unsigned)((totX + 255) / 256), 256, 0, stream>>>(x, xbf, totX);

    dim3 pg0((KPAD0 + 255) / 256, G4H);
    prep_weight_kernel<<<pg0, 256, 0, stream>>>(Wih0f, bih0f, bhh0f, w0f, bias0f, IN_, KPAD0);
    prep_weight_kernel<<<pg0, 256, 0, stream>>>(Wih0b, bih0b, bhh0b, w0b, bias0b, IN_, KPAD0);
    dim3 pg1(2, G4H);
    prep_weight_kernel<<<pg1, 256, 0, stream>>>(Wih1f, bih1f, bhh1f, w1f, bias1f, 512, 512);
    prep_weight_kernel<<<pg1, 256, 0, stream>>>(Wih1b, bih1b, bhh1b, w1b, bias1b, 512, 512);
    dim3 pgh(1, G4H);
    prep_weight_kernel<<<pgh, 256, 0, stream>>>(Whh0f, nullptr, nullptr, whh0f, nullptr, H_, H_);
    prep_weight_kernel<<<pgh, 256, 0, stream>>>(Whh0b, nullptr, nullptr, whh0b, nullptr, H_, H_);
    prep_weight_kernel<<<pgh, 256, 0, stream>>>(Whh1f, nullptr, nullptr, whh1f, nullptr, H_, H_);
    prep_weight_kernel<<<pgh, 256, 0, stream>>>(Whh1b, nullptr, nullptr, whh1b, nullptr, H_, H_);

    dim3 gg(MROWS / 128, G4H / 64);   // 128 x 16
    // layer 0 input projections
    gemm_bf16_kernel<<<gg, 256, 0, stream>>>(xbf, w0f, bias0f, preF, KPAD0);
    gemm_bf16_kernel<<<gg, 256, 0, stream>>>(xbf, w0b, bias0b, preB, KPAD0);
    // layer 0 scan -> h1 (bf16, fwd cols 0-255 / bwd cols 256-511)
    lstm_scan_kernel<<<2, 512, 0, stream>>>(preF, preB, whh0f, whh0b, h1bf, nullptr);
    // layer 1 input projections (reuse pre buffers)
    gemm_bf16_kernel<<<gg, 256, 0, stream>>>(h1bf, w1f, bias1f, preF, 512);
    gemm_bf16_kernel<<<gg, 256, 0, stream>>>(h1bf, w1b, bias1b, preB, 512);
    // layer 1 scan -> h2 (f32)
    lstm_scan_kernel<<<2, 512, 0, stream>>>(preF, preB, whh1f, whh1b, nullptr, h2);

    int totO = MROWS * NCLS;
    fc_kernel<<<(totO + 255) / 256, 256, 0, stream>>>(
        h2, Wfc, bfc, (float*)d_out + (size_t)MROWS * IN_);
}
